// EfficientCrossAttentionHead3modals_489626271901
// MI455X (gfx1250) — matrix operand order repaired and checked by reading, not compiled, not run on hardware
//
#include <hip/hip_runtime.h>
#include <hip/hip_bf16.h>

typedef __attribute__((ext_vector_type(16))) __bf16 v16bf;
typedef __attribute__((ext_vector_type(8)))  float  v8f;

// Problem constants
constexpr int BATCH = 4;
constexpr int NTOK  = 8192;
constexpr int NE    = 512;
constexpr int HD    = 64;
constexpr int ROWS  = 128;              // rows per workgroup (8 waves x 16)
constexpr int NBLK  = NTOK / ROWS;      // 64
constexpr int PSTR  = HD + HD + HD*HD;  // per-block partial stride (floats) = 4224

// ---------------------------------------------------------------- helpers

__device__ inline v8f wmma_bf16(v16bf a, v16bf b, v8f c) {
  // D = A*B + C, 16x16x32 bf16 -> f32
  return __builtin_amdgcn_wmma_f32_16x16x32_bf16(
      false, a, false, b, (short)0, c, false, false);
}

// A-fragment (16x32) from fp32 row-major source; lane holds row = lane&15.
// K elements: {kb..kb+7, kb+16..kb+23}, kb = (lane>=16) ? 8 : 0.
__device__ inline v16bf a_frag_f32(const float* __restrict__ rowp, int kc, int lane) {
  const int kb = (lane & 16) ? 8 : 0;
  const float* p = rowp + kc + kb;
  v16bf r;
#pragma unroll
  for (int j = 0; j < 8; ++j) {
    r[j]     = (__bf16)p[j];
    r[j + 8] = (__bf16)p[j + 16];
  }
  return r;
}

// A-fragment (16x32) from bf16 row-major source (same K pattern).
__device__ inline v16bf a_frag_bf16(const __bf16* rowp, int kc, int lane) {
  const int kb = (lane & 16) ? 8 : 0;
  const __bf16* p = rowp + kc + kb;
  v16bf r;
#pragma unroll
  for (int j = 0; j < 8; ++j) {
    r[j]     = p[j];
    r[j + 8] = p[j + 16];
  }
  return r;
}

// B-fragment (32x16): lane holds column; K elements:
// {kb..kb+7, kb+16..kb+23}, kb = (lane>=16) ? 8 : 0. colp points to that column's K-row.
__device__ inline v16bf b_frag_bf16(const __bf16* colp, int kc, int lane) {
  typedef __attribute__((ext_vector_type(8))) __bf16 v8bf;
  const int kb = (lane & 16) ? 8 : 0;
  v8bf lo = *(const v8bf*)(colp + kc + kb);
  v8bf hi = *(const v8bf*)(colp + kc + kb + 16);
  return __builtin_shufflevector(lo, hi, 0,1,2,3,4,5,6,7,8,9,10,11,12,13,14,15);
}

__device__ inline void atomicMaxF(float* addr, float val) {
  int* ia = (int*)addr;
  int old = __float_as_int(*addr);
  while (__int_as_float(old) < val) {
    int assumed = old;
    old = atomicCAS(ia, assumed, __float_as_int(val));
    if (old == assumed) break;
  }
}

// ------------------------------------------------- kernel 0: weight convert
// wT slots: [0]=Wk0 [1]=Wq0 [2]=Wv0 [3]=Wk1 [4]=Wq1 [5]=Wv1, each [d][k] bf16
__global__ __launch_bounds__(256) void k_wconv(
    const float* __restrict__ s0, const float* __restrict__ s1,
    const float* __restrict__ s2, const float* __restrict__ s3,
    const float* __restrict__ s4, const float* __restrict__ s5,
    __bf16* __restrict__ wT) {
  int t = blockIdx.x * 256 + threadIdx.x;            // < 6*512*64
  int mat = t >> 15;
  int i   = t & 32767;
  int k = i >> 6, d = i & 63;
  const float* src;
  switch (mat) {
    case 0: src = s0; break; case 1: src = s1; break; case 2: src = s2; break;
    case 3: src = s3; break; case 4: src = s4; break; default: src = s5; break;
  }
  wT[(size_t)mat * (NE * HD) + d * NE + k] = (__bf16)src[k * HD + d];
}

// ------------------------------------------------- kernel 1: projections
// Per workgroup: (stream s, batch b, row-block blk). 8 waves x 16 rows.
// Computes K/Q/V via WMMA, fuses K-column stats + partial E^T*V (WMMA) and
// Q row-softmax (written into d_out as temporary q storage).
__global__ __launch_bounds__(256) void k_proj(
    const float* __restrict__ x0, const float* __restrict__ x1,
    const float* __restrict__ x2, const __bf16* __restrict__ wT,
    float* __restrict__ qout, float* __restrict__ partials) {
  const int tid  = threadIdx.x;
  const int lane = tid & 31;
  const int wave = tid >> 5;
  const int nl   = lane & 15;
  const int grp  = (lane >> 4) & 1;

  const int bid = blockIdx.x;
  const int s   = bid / (BATCH * NBLK);
  const int rem = bid % (BATCH * NBLK);
  const int b   = rem / NBLK;
  const int blk = rem % NBLK;

  const float* xs = (s == 0) ? x0 : ((s == 1) ? x1 : x2);
  const int set = (s == 0) ? 0 : 1;
  const __bf16* wK = wT + (size_t)(set * 3 + 0) * (NE * HD);
  const __bf16* wQ = wT + (size_t)(set * 3 + 1) * (NE * HD);
  const __bf16* wV = wT + (size_t)(set * 3 + 2) * (NE * HD);

  const int rowg = blk * ROWS + wave * 16 + nl;      // this lane's A row
  const float* xrow = xs + ((size_t)b * NTOK + rowg) * NE;

  v8f accK[4], accQ[4], accV[4];
#pragma unroll
  for (int c = 0; c < 4; ++c) { accK[c] = (v8f){}; accQ[c] = (v8f){}; accV[c] = (v8f){}; }

  // main GEMM: 16 K-steps of 32, 12 WMMAs per step.
  // Batch all 24 b128 B-loads before the 12 WMMAs so s_wait_loadcnt counts
  // stagger and B-load latency overlaps with matrix issue.
  for (int kc = 0; kc < NE; kc += 32) {
    v16bf a = a_frag_f32(xrow, kc, lane);
    v16bf bk[4], bq[4], bv[4];
#pragma unroll
    for (int c = 0; c < 4; ++c) {
      const int d = c * 16 + nl;
      bk[c] = b_frag_bf16(wK + (size_t)d * NE, kc, lane);
      bq[c] = b_frag_bf16(wQ + (size_t)d * NE, kc, lane);
      bv[c] = b_frag_bf16(wV + (size_t)d * NE, kc, lane);
    }
#pragma unroll
    for (int c = 0; c < 4; ++c) {
      accK[c] = wmma_bf16(a, bk[c], accK[c]);
      accQ[c] = wmma_bf16(a, bq[c], accQ[c]);
      accV[c] = wmma_bf16(a, bv[c], accV[c]);
    }
  }

  // ---------------- fused epilogue ----------------
  __shared__ float  qbuf[ROWS * HD];     // 32 KB, fp32 q staging
  __shared__ __bf16 eT[HD * ROWS];       // 16 KB, E^T = exp(k - m)  [h][n]
  __shared__ __bf16 vT[HD * ROWS];       // 16 KB, V^T               [d][n]
  __shared__ float  cmax[HD];
  __shared__ float  csum[HD];

  if (tid < HD) { cmax[tid] = -1e30f; csum[tid] = 0.f; }
  __syncthreads();

  // K column max (over this block's 128 rows)
#pragma unroll
  for (int c = 0; c < 4; ++c) {
    float m = accK[c][0];
#pragma unroll
    for (int r = 1; r < 8; ++r) m = fmaxf(m, accK[c][r]);
    atomicMaxF(&cmax[c * 16 + nl], m);
  }
  __syncthreads();

  const int rl = wave * 16;              // local row base of this wave
#pragma unroll
  for (int c = 0; c < 4; ++c) {
    const int h  = c * 16 + nl;
    const float mv = cmax[h];
    float part = 0.f;
#pragma unroll
    for (int r = 0; r < 8; ++r) {
      const int n = rl + r + 8 * grp;
      float e = __expf(accK[c][r] - mv);
      part += e;
      eT[h * ROWS + n] = (__bf16)e;
      vT[h * ROWS + n] = (__bf16)accV[c][r];   // h doubles as d index here
      qbuf[n * HD + h] = accQ[c][r];
    }
    atomicAdd(&csum[h], part);
  }
  __syncthreads();

  // Q row-softmax -> qout (temporary storage inside d_out, later overwritten)
  if (tid < ROWS) {
    float* qr = qbuf + tid * HD;
    float m = qr[0];
#pragma unroll 8
    for (int j = 1; j < HD; ++j) m = fmaxf(m, qr[j]);
    float ssum = 0.f;
#pragma unroll 8
    for (int j = 0; j < HD; ++j) { float e = __expf(qr[j] - m); qr[j] = e; ssum += e; }
    const float inv = 1.f / ssum;
    float* dst = qout + (((size_t)s * BATCH + b) * NTOK + blk * ROWS + tid) * HD;
#pragma unroll 8
    for (int j = 0; j < HD; ++j) dst[j] = qr[j] * inv;
  }

  // partial W = E^T * V  (64x64), via WMMA over LDS tiles; 2 tiles per wave
  float* pbase = partials + (size_t)bid * PSTR;
#pragma unroll
  for (int t = wave * 2; t < wave * 2 + 2; ++t) {
    const int th = t >> 2, td = t & 3;
    v8f acc = (v8f){};
    v16bf ea[4], vb[4];
#pragma unroll
    for (int kk = 0; kk < 4; ++kk) {
      ea[kk] = a_frag_bf16(eT + (th * 16 + nl) * ROWS, kk * 32, lane);
      vb[kk] = b_frag_bf16(vT + (td * 16 + nl) * ROWS, kk * 32, lane);
    }
#pragma unroll
    for (int kk = 0; kk < 4; ++kk) acc = wmma_bf16(ea[kk], vb[kk], acc);
#pragma unroll
    for (int r = 0; r < 8; ++r) {
      const int h = th * 16 + r + 8 * grp;
      const int d = td * 16 + nl;
      pbase[2 * HD + h * HD + d] = acc[r];
    }
  }
  if (tid < HD) { pbase[tid] = cmax[tid]; pbase[HD + tid] = csum[tid]; }
}

// ------------------------------------------------- kernel 2: combine partials
__global__ __launch_bounds__(256) void k_combine(
    const float* __restrict__ partials, float* __restrict__ wmat) {
  const int sb  = blockIdx.x;            // s*BATCH + b, 0..11
  const int tid = threadIdx.x;
  const float* base = partials + (size_t)sb * NBLK * PSTR;
  __shared__ float M[HD], S[HD];

  if (tid < HD) {
    float m = -1e30f;
    for (int blk = 0; blk < NBLK; ++blk)
      m = fmaxf(m, base[(size_t)blk * PSTR + tid]);
    float ssum = 0.f;
    for (int blk = 0; blk < NBLK; ++blk)
      ssum += __expf(base[(size_t)blk * PSTR + tid] - m) *
              base[(size_t)blk * PSTR + HD + tid];
    M[tid] = m; S[tid] = ssum;
  }
  __syncthreads();

  for (int i = tid; i < HD * HD; i += 256) {
    const int h = i >> 6;
    float acc = 0.f;
    const float mh = M[h];
    for (int blk = 0; blk < NBLK; ++blk) {
      const float* pb = base + (size_t)blk * PSTR;
      acc += __expf(pb[h] - mh) * pb[2 * HD + i];
    }
    wmat[(size_t)sb * (HD * HD) + i] = acc / S[h];
  }
}

// ------------------------------------------------- kernel 3: attn = q @ (wa+wb)
__global__ __launch_bounds__(256) void k_attn(
    const float* __restrict__ wmat, float* __restrict__ out) {
  const int tid  = threadIdx.x;
  const int lane = tid & 31;
  const int wave = tid >> 5;
  const int nl   = lane & 15;
  const int grp  = (lane >> 4) & 1;

  const int bid = blockIdx.x;
  const int s   = bid / (BATCH * NBLK);
  const int rem = bid % (BATCH * NBLK);
  const int b   = rem / NBLK;
  const int blk = rem % NBLK;

  const int p0 = (s == 0) ? 1 : 0;
  const int p1 = (s == 2) ? 1 : 2;
  const float* wa = wmat + ((size_t)p0 * BATCH + b) * (HD * HD);
  const float* wb = wmat + ((size_t)p1 * BATCH + b) * (HD * HD);

  __shared__ __bf16 wT2[HD * HD];        // (wa+wb)^T  [d][h]
  for (int i = tid; i < HD * HD; i += 256) {
    const int h = i >> 6, d = i & 63;
    wT2[d * HD + h] = (__bf16)(wa[i] + wb[i]);
  }
  __syncthreads();

  const int rowg = blk * ROWS + wave * 16 + nl;
  const float* qrow = out + (((size_t)s * BATCH + b) * NTOK + rowg) * HD;

  v8f acc[4];
#pragma unroll
  for (int c = 0; c < 4; ++c) acc[c] = (v8f){};

#pragma unroll
  for (int kc = 0; kc < HD; kc += 32) {
    v16bf a = a_frag_f32(qrow, kc, lane);
    v16bf bb[4];
#pragma unroll
    for (int c = 0; c < 4; ++c)
      bb[c] = b_frag_bf16(wT2 + (c * 16 + nl) * HD, kc, lane);
#pragma unroll
    for (int c = 0; c < 4; ++c)
      acc[c] = wmma_bf16(a, bb[c], acc[c]);
  }

  const size_t obase = ((size_t)s * BATCH + b) * NTOK + blk * ROWS + wave * 16;
#pragma unroll
  for (int c = 0; c < 4; ++c)
#pragma unroll
    for (int r = 0; r < 8; ++r)
      out[(obase + r + 8 * grp) * HD + c * 16 + nl] = acc[c][r];
}

// ------------------------------------------------- launcher
extern "C" void kernel_launch(void* const* d_in, const int* in_sizes, int n_in,
                              void* d_out, int out_size, void* d_ws, size_t ws_size,
                              hipStream_t stream) {
  (void)in_sizes; (void)n_in; (void)out_size; (void)ws_size;
  const float* x0  = (const float*)d_in[0];
  const float* x1  = (const float*)d_in[1];
  const float* x2  = (const float*)d_in[2];
  const float* Wk0 = (const float*)d_in[3];
  const float* Wk1 = (const float*)d_in[4];
  const float* Wq0 = (const float*)d_in[5];
  const float* Wq1 = (const float*)d_in[6];
  const float* Wv0 = (const float*)d_in[7];
  const float* Wv1 = (const float*)d_in[8];
  float* out = (float*)d_out;

  // workspace layout
  __bf16* wT      = (__bf16*)d_ws;                               // 6*64*512 bf16
  float*  part    = (float*)((char*)d_ws + (size_t)6 * NE * HD * sizeof(__bf16));
  float*  wmat    = part + (size_t)3 * BATCH * NBLK * PSTR;      // 12*4096 f32

  const int NWG = 3 * BATCH * NBLK;  // 768

  k_wconv<<<(6 * NE * HD) / 256, 256, 0, stream>>>(Wk0, Wq0, Wv0, Wk1, Wq1, Wv1, wT);
  k_proj<<<NWG, 256, 0, stream>>>(x0, x1, x2, wT, out, part);
  k_combine<<<3 * BATCH, 256, 0, stream>>>(part, wmat);
  k_attn<<<NWG, 256, 0, stream>>>(wmat, out);
}
